// CausalRecurrenceLayer_2680059593237
// MI455X (gfx1250) — matrix operand order repaired and checked by reading, not compiled, not run on hardware
//
#include <hip/hip_runtime.h>
#include <math.h>

typedef __attribute__((ext_vector_type(16))) __bf16 v16bf;
typedef __attribute__((ext_vector_type(8)))  __bf16 bf16x8;
typedef __attribute__((ext_vector_type(8)))  float  v8f;
typedef int v4i __attribute__((vector_size(16)));   // matches async-LDS builtin param type

#define DCH   1024
#define BATCH 4
#define SEQ   8192
#define MTOT  (BATCH*SEQ)   // 32768 tokens
#define NCH   32            // scan chunks
#define CHL   (SEQ/NCH)     // 256 steps per chunk

#define BM 64
#define BN 128
#define BK 32
#define LDA 40   // padded halves per row (80B: keeps 16B alignment, staggers banks)
#define LDB 40

#if defined(__HIP_DEVICE_COMPILE__)
#if __has_builtin(__builtin_amdgcn_global_load_async_to_lds_b128) && \
    __has_builtin(__builtin_amdgcn_s_wait_asynccnt)
#define USE_ASYNC_LDS 1
#endif
#endif

union Frag { v16bf v; bf16x8 h[2]; };

// 16-byte global -> LDS copy; async DMA (ASYNCcnt) on gfx1250 when available.
__device__ __forceinline__ void cp_b128(void* lds, const void* g) {
#ifdef USE_ASYNC_LDS
  __builtin_amdgcn_global_load_async_to_lds_b128(
      (__attribute__((address_space(1))) v4i*)g,
      (__attribute__((address_space(3))) v4i*)lds, 0, 0);
#else
  *(bf16x8*)lds = *(const bf16x8*)g;
#endif
}
__device__ __forceinline__ void cp_wait() {
#ifdef USE_ASYNC_LDS
  __builtin_amdgcn_s_wait_asynccnt(0);
#endif
}

// ---------------- depthwise causal conv: x -> xc (f32) + xcb (bf16) ----------------
__global__ void conv_kernel(const float* __restrict__ x, const float* __restrict__ w,
                            const float* __restrict__ bias,
                            float* __restrict__ xc, __bf16* __restrict__ xcb) {
  long idx = (long)blockIdx.x * blockDim.x + threadIdx.x;   // < MTOT*DCH
  int c = (int)(idx % DCH);
  long bl = idx / DCH;
  int l = (int)(bl % SEQ);
  float acc = bias[c];
#pragma unroll
  for (int k = 0; k < 4; ++k) {
    int ls = l - 3 + k;
    if (ls >= 0) acc += x[idx + (long)(ls - l) * DCH] * w[c * 4 + k];
  }
  xc[idx] = acc;
  xcb[idx] = (__bf16)acc;
}

// ---------------- weight bf16 conversion + a-coefficient ----------------
__global__ void prep_kernel(const float* __restrict__ wr, const float* __restrict__ wi,
                            const float* __restrict__ wo, const float* __restrict__ log_a,
                            __bf16* __restrict__ wrb, __bf16* __restrict__ wib,
                            __bf16* __restrict__ wob, float* __restrict__ coef) {
  int idx = blockIdx.x * 256 + threadIdx.x;   // < DCH*DCH
  wrb[idx] = (__bf16)wr[idx];
  wib[idx] = (__bf16)wi[idx];
  wob[idx] = (__bf16)wo[idx];
  if (idx < DCH) {
    float s = 1.0f / (1.0f + expf(-log_a[idx]));
    coef[idx] = 8.0f * logf(fmaxf(s, 1e-30f));   // a_t = exp(r * coef)
  }
}

// ---------------- fused dual GEMM + gate epilogue ----------------
__global__ __launch_bounds__(256) void gates_kernel(
    const __bf16* __restrict__ A,  const __bf16* __restrict__ Br, const __bf16* __restrict__ Bi,
    const float* __restrict__ rbias, const float* __restrict__ ibias,
    const float* __restrict__ coef,  const float* __restrict__ xc,
    float* __restrict__ a_out, float* __restrict__ bt_out) {
  __shared__ __bf16 As[2][BM * LDA];
  __shared__ __bf16 Bs[2][2][BN * LDB];
  int t = threadIdx.x;
  int lane = t & 31, wave = t >> 5;
  int wm = wave >> 2, wn = wave & 3;        // 2 x 4 wave grid
  int half = lane >> 4, lq = lane & 15;
  long m0 = (long)blockIdx.y * BM;
  int  n0 = blockIdx.x * BN;

  v8f zero = {0.f,0.f,0.f,0.f,0.f,0.f,0.f,0.f};
  v8f accR[2][2], accI[2][2];
#pragma unroll
  for (int a2 = 0; a2 < 2; ++a2)
#pragma unroll
    for (int b2 = 0; b2 < 2; ++b2) { accR[a2][b2] = zero; accI[a2][b2] = zero; }

  int ar = t >> 2, ac = (t & 3) * 8;        // A staging: 64 rows x 4 chunks of 8 halves
  int brr = t >> 1, bc = (t & 1) * 16;      // B staging: 128 rows x 2 threads/row

  const __bf16* gA  = &A[(m0 + ar) * DCH + ac];
  const __bf16* gBr = &Br[(long)(n0 + brr) * DCH + bc];
  const __bf16* gBi = &Bi[(long)(n0 + brr) * DCH + bc];

  // stage chunk 0 into buffer 0
  cp_b128(&As[0][ar * LDA + ac],           gA);
  cp_b128(&Bs[0][0][brr * LDB + bc],       gBr);
  cp_b128(&Bs[0][0][brr * LDB + bc + 8],   gBr + 8);
  cp_b128(&Bs[0][1][brr * LDB + bc],       gBi);
  cp_b128(&Bs[0][1][brr * LDB + bc + 8],   gBi + 8);
  cp_wait();
  __syncthreads();

  for (int k0 = 0; k0 < DCH; k0 += BK) {
    int buf = (k0 / BK) & 1;
    int kn = k0 + BK;
    if (kn < DCH) {   // issue next chunk's DMA before consuming this one
      cp_b128(&As[buf ^ 1][ar * LDA + ac],         gA + kn);
      cp_b128(&Bs[buf ^ 1][0][brr * LDB + bc],     gBr + kn);
      cp_b128(&Bs[buf ^ 1][0][brr * LDB + bc + 8], gBr + kn + 8);
      cp_b128(&Bs[buf ^ 1][1][brr * LDB + bc],     gBi + kn);
      cp_b128(&Bs[buf ^ 1][1][brr * LDB + bc + 8], gBi + kn + 8);
#ifndef USE_ASYNC_LDS
      if (kn + BK < DCH) {
        __builtin_prefetch(gA + kn + BK, 0, 1);
        __builtin_prefetch(gBr + kn + BK, 0, 1);
        __builtin_prefetch(gBi + kn + BK, 0, 1);
      }
#endif
    }

    Frag fa[2], fbr[2], fbi[2];
#pragma unroll
    for (int ms = 0; ms < 2; ++ms) {
      int row = 32 * wm + 16 * ms + lq;
      fa[ms].h[0] = *(const bf16x8*)&As[buf][row * LDA + 16 * half];
      fa[ms].h[1] = *(const bf16x8*)&As[buf][row * LDA + 16 * half + 8];
    }
#pragma unroll
    for (int ns = 0; ns < 2; ++ns) {
      int row = 32 * wn + 16 * ns + lq;
      fbr[ns].h[0] = *(const bf16x8*)&Bs[buf][0][row * LDB + 16 * half];
      fbr[ns].h[1] = *(const bf16x8*)&Bs[buf][0][row * LDB + 16 * half + 8];
      fbi[ns].h[0] = *(const bf16x8*)&Bs[buf][1][row * LDB + 16 * half];
      fbi[ns].h[1] = *(const bf16x8*)&Bs[buf][1][row * LDB + 16 * half + 8];
    }
#pragma unroll
    for (int ms = 0; ms < 2; ++ms)
#pragma unroll
      for (int ns = 0; ns < 2; ++ns) {
        accR[ms][ns] = __builtin_amdgcn_wmma_f32_16x16x32_bf16(
            false, fa[ms].v, false, fbr[ns].v, (short)0, accR[ms][ns], false, false);
        accI[ms][ns] = __builtin_amdgcn_wmma_f32_16x16x32_bf16(
            false, fa[ms].v, false, fbi[ns].v, (short)0, accI[ms][ns], false, false);
      }
    cp_wait();          // next chunk's DMA landed in LDS
    __syncthreads();    // all waves done reading buf / writing buf^1
  }

#pragma unroll
  for (int ms = 0; ms < 2; ++ms)
#pragma unroll
    for (int ns = 0; ns < 2; ++ns) {
      int n = n0 + 32 * wn + 16 * ns + lq;
      float br = rbias[n], bi = ibias[n], cf = coef[n];
#pragma unroll
      for (int j = 0; j < 8; ++j) {
        long m = m0 + 32 * wm + 16 * ms + 8 * half + j;
        float r = 1.f / (1.f + expf(-(accR[ms][ns][j] + br)));
        float i = 1.f / (1.f + expf(-(accI[ms][ns][j] + bi)));
        float a = expf(r * cf);
        float xcv = xc[m * DCH + n];
        float bt = sqrtf(fmaxf(1.f - a * a, 1e-6f)) * (i * xcv);
        a_out[m * DCH + n]  = a;
        bt_out[m * DCH + n] = bt;
      }
    }
}

// ---------------- chunked linear-recurrence scan (h = a*h + b) ----------------
__global__ void scan1_kernel(const float* __restrict__ a, const float* __restrict__ bt,
                             float* __restrict__ Ap, float* __restrict__ He) {
  int c = blockIdx.x * 256 + threadIdx.x;
  int ch = blockIdx.y, b = blockIdx.z;
  long base = ((long)b * SEQ + (long)ch * CHL) * DCH + c;
  float Aprod = 1.f, h = 0.f;
  for (int l = 0; l < CHL; ++l) {
    float av = a[base + (long)l * DCH];
    float bv = bt[base + (long)l * DCH];
    h = fmaf(av, h, bv);
    Aprod *= av;
  }
  long ci = ((long)b * NCH + ch) * DCH + c;
  Ap[ci] = Aprod; He[ci] = h;
}

__global__ void scan2_kernel(const float* __restrict__ Ap, const float* __restrict__ He,
                             float* __restrict__ Cin) {
  int idx = blockIdx.x * 256 + threadIdx.x;   // < BATCH*DCH
  int b = idx / DCH, c = idx % DCH;
  float h = 0.f;
  for (int ch = 0; ch < NCH; ++ch) {
    long ci = ((long)b * NCH + ch) * DCH + c;
    Cin[ci] = h;
    h = fmaf(Ap[ci], h, He[ci]);
  }
}

__global__ void scan3_kernel(const float* __restrict__ a, const float* __restrict__ bt,
                             const float* __restrict__ Cin, __bf16* __restrict__ hb) {
  int c = blockIdx.x * 256 + threadIdx.x;
  int ch = blockIdx.y, b = blockIdx.z;
  long base = ((long)b * SEQ + (long)ch * CHL) * DCH + c;
  float h = Cin[((long)b * NCH + ch) * DCH + c];
  for (int l = 0; l < CHL; ++l) {
    float av = a[base + (long)l * DCH];
    float bv = bt[base + (long)l * DCH];
    h = fmaf(av, h, bv);
    hb[base + (long)l * DCH] = (__bf16)h;
  }
}

// ---------------- output GEMM: y = h @ out_w^T ----------------
__global__ __launch_bounds__(256) void out_kernel(const __bf16* __restrict__ A,
                                                  const __bf16* __restrict__ Bw,
                                                  float* __restrict__ y) {
  __shared__ __bf16 As[2][BM * LDA];
  __shared__ __bf16 Bs[2][BN * LDB];
  int t = threadIdx.x;
  int lane = t & 31, wave = t >> 5;
  int wm = wave >> 2, wn = wave & 3;
  int half = lane >> 4, lq = lane & 15;
  long m0 = (long)blockIdx.y * BM;
  int  n0 = blockIdx.x * BN;

  v8f zero = {0.f,0.f,0.f,0.f,0.f,0.f,0.f,0.f};
  v8f acc[2][2];
#pragma unroll
  for (int a2 = 0; a2 < 2; ++a2)
#pragma unroll
    for (int b2 = 0; b2 < 2; ++b2) acc[a2][b2] = zero;

  int ar = t >> 2, ac = (t & 3) * 8;
  int brr = t >> 1, bc = (t & 1) * 16;

  const __bf16* gA = &A[(m0 + ar) * DCH + ac];
  const __bf16* gB = &Bw[(long)(n0 + brr) * DCH + bc];

  cp_b128(&As[0][ar * LDA + ac],         gA);
  cp_b128(&Bs[0][brr * LDB + bc],        gB);
  cp_b128(&Bs[0][brr * LDB + bc + 8],    gB + 8);
  cp_wait();
  __syncthreads();

  for (int k0 = 0; k0 < DCH; k0 += BK) {
    int buf = (k0 / BK) & 1;
    int kn = k0 + BK;
    if (kn < DCH) {
      cp_b128(&As[buf ^ 1][ar * LDA + ac],      gA + kn);
      cp_b128(&Bs[buf ^ 1][brr * LDB + bc],     gB + kn);
      cp_b128(&Bs[buf ^ 1][brr * LDB + bc + 8], gB + kn + 8);
#ifndef USE_ASYNC_LDS
      if (kn + BK < DCH) {
        __builtin_prefetch(gA + kn + BK, 0, 1);
        __builtin_prefetch(gB + kn + BK, 0, 1);
      }
#endif
    }

    Frag fa[2], fb[2];
#pragma unroll
    for (int ms = 0; ms < 2; ++ms) {
      int row = 32 * wm + 16 * ms + lq;
      fa[ms].h[0] = *(const bf16x8*)&As[buf][row * LDA + 16 * half];
      fa[ms].h[1] = *(const bf16x8*)&As[buf][row * LDA + 16 * half + 8];
    }
#pragma unroll
    for (int ns = 0; ns < 2; ++ns) {
      int row = 32 * wn + 16 * ns + lq;
      fb[ns].h[0] = *(const bf16x8*)&Bs[buf][row * LDB + 16 * half];
      fb[ns].h[1] = *(const bf16x8*)&Bs[buf][row * LDB + 16 * half + 8];
    }
#pragma unroll
    for (int ms = 0; ms < 2; ++ms)
#pragma unroll
      for (int ns = 0; ns < 2; ++ns)
        acc[ms][ns] = __builtin_amdgcn_wmma_f32_16x16x32_bf16(
            false, fa[ms].v, false, fb[ns].v, (short)0, acc[ms][ns], false, false);
    cp_wait();
    __syncthreads();
  }

#pragma unroll
  for (int ms = 0; ms < 2; ++ms)
#pragma unroll
    for (int ns = 0; ns < 2; ++ns) {
      int n = n0 + 32 * wn + 16 * ns + lq;
#pragma unroll
      for (int j = 0; j < 8; ++j) {
        long m = m0 + 32 * wm + 16 * ms + 8 * half + j;
        y[m * DCH + n] = acc[ms][ns][j];
      }
    }
}

// ---------------- RMSNorm ----------------
__global__ void rmsnorm_kernel(const float* __restrict__ y, const float* __restrict__ nw,
                               float* __restrict__ out) {
  __shared__ float red[256];
  long row = blockIdx.x;
  const float* yr = y + row * DCH;
  float s = 0.f;
  for (int n = threadIdx.x; n < DCH; n += 256) { float v = yr[n]; s += v * v; }
  red[threadIdx.x] = s;
  __syncthreads();
  for (int o = 128; o > 0; o >>= 1) {
    if ((int)threadIdx.x < o) red[threadIdx.x] += red[threadIdx.x + o];
    __syncthreads();
  }
  float scale = rsqrtf(red[0] * (1.0f / DCH) + 1e-6f);
  for (int n = threadIdx.x; n < DCH; n += 256)
    out[row * DCH + n] = yr[n] * scale * nw[n];
}

extern "C" void kernel_launch(void* const* d_in, const int* in_sizes, int n_in,
                              void* d_out, int out_size, void* d_ws, size_t ws_size,
                              hipStream_t stream) {
  const float* x  = (const float*)d_in[0];
  const float* cw = (const float*)d_in[1];
  const float* cb = (const float*)d_in[2];
  const float* Wr = (const float*)d_in[3];
  const float* rb = (const float*)d_in[4];
  const float* Wi = (const float*)d_in[5];
  const float* ib = (const float*)d_in[6];
  const float* la = (const float*)d_in[7];
  const float* Wo = (const float*)d_in[8];
  const float* nw = (const float*)d_in[9];
  float* out = (float*)d_out;

  char* ws = (char*)d_ws;
  size_t off = 0;
  auto alloc = [&](size_t bytes) -> void* {
    void* p = ws + off;
    off = (off + bytes + 255) & ~(size_t)255;
    return p;
  };

  float*  xc     = (float*)alloc((size_t)MTOT * DCH * 4);      // aliased as y later
  __bf16* xcb    = (__bf16*)alloc((size_t)MTOT * DCH * 2);     // aliased as h_bf16 later
  __bf16* wrb    = (__bf16*)alloc((size_t)DCH * DCH * 2);
  __bf16* wib    = (__bf16*)alloc((size_t)DCH * DCH * 2);
  __bf16* wob    = (__bf16*)alloc((size_t)DCH * DCH * 2);
  float*  coef   = (float*)alloc((size_t)DCH * 4);
  float*  a_arr  = (float*)alloc((size_t)MTOT * DCH * 4);
  float*  bt_arr = (float*)alloc((size_t)MTOT * DCH * 4);
  float*  Ap     = (float*)alloc((size_t)BATCH * NCH * DCH * 4);
  float*  He     = (float*)alloc((size_t)BATCH * NCH * DCH * 4);
  float*  Cin    = (float*)alloc((size_t)BATCH * NCH * DCH * 4);
  float*  y      = xc;    // xc dead after gates_kernel
  __bf16* hb     = xcb;   // xcb dead after gates_kernel

  conv_kernel<<<(MTOT * DCH) / 256, 256, 0, stream>>>(x, cw, cb, xc, xcb);
  prep_kernel<<<(DCH * DCH) / 256, 256, 0, stream>>>(Wr, Wi, Wo, la, wrb, wib, wob, coef);
  gates_kernel<<<dim3(DCH / BN, MTOT / BM), 256, 0, stream>>>(
      xcb, wrb, wib, rb, ib, coef, xc, a_arr, bt_arr);
  scan1_kernel<<<dim3(DCH / 256, NCH, BATCH), 256, 0, stream>>>(a_arr, bt_arr, Ap, He);
  scan2_kernel<<<(BATCH * DCH) / 256, 256, 0, stream>>>(Ap, He, Cin);
  scan3_kernel<<<dim3(DCH / 256, NCH, BATCH), 256, 0, stream>>>(a_arr, bt_arr, Cin, hb);
  out_kernel<<<dim3(DCH / BN, MTOT / BM), 256, 0, stream>>>(hb, wob, y);
  rmsnorm_kernel<<<MTOT, 256, 0, stream>>>(y, nw, out);
}
